// GNNGlobal_42460046688959
// MI455X (gfx1250) — matrix operand order repaired and checked by reading, not compiled, not run on hardware
//
#include <hip/hip_runtime.h>
#include <hip/hip_bf16.h>

#define Bb 64
#define Nn 1024
#define Ee 16384
#define Hh 64
#define Vv 128

typedef __attribute__((ext_vector_type(16))) __bf16 bf16x16;
typedef __attribute__((ext_vector_type(8)))  float  f32x8;

union FragU { bf16x16 v; unsigned int w[8]; };

// ---- shared WMMA fragment loaders (contiguous k order, A and B alike) ----
// A (16x32 bf16): lanes 0-15 -> M=lane, K 0..15 ; lanes 16-31 -> M=lane-16, K 16..31
// rowp points at this lane's M row + k-block start; h2 = 0 (lanes<16) or 4 (lanes>=16), in uint units
__device__ inline bf16x16 load_a_frag(const unsigned short* rowp, int h2) {
  FragU f;
  const unsigned int* p = (const unsigned int*)rowp;
#pragma unroll
  for (int i = 0; i < 4; i++) { f.w[i] = p[2 * h2 + i]; f.w[4 + i] = p[2 * h2 + 4 + i]; }
  return f.v;
}
// B (32x16 bf16): lane -> N column; lanes<16 hold K 0..15, lanes>=16 hold K 16..31 (contiguous in wT[n][k])
__device__ inline bf16x16 load_b_frag(const unsigned short* colp) {
  FragU f;
  const unsigned int* p = (const unsigned int*)colp;
#pragma unroll
  for (int i = 0; i < 8; i++) f.w[i] = p[i];
  return f.v;
}

// ---------------- utility kernels ----------------

__global__ void fill0_kernel(float* p, size_t count) {
  size_t t = (size_t)blockIdx.x * 256 + threadIdx.x;
  if (t < count) p[t] = 0.f;
}

__global__ void embed_kernel(const int* __restrict__ idx, const float* __restrict__ emb,
                             float* __restrict__ x, size_t nodes) {
  size_t t = (size_t)blockIdx.x * 256 + threadIdx.x;
  if (t >= nodes * Hh) return;
  size_t node = t >> 6;
  int h = (int)(t & 63);
  x[t] = emb[(size_t)idx[node] * Hh + h];
}

__global__ void edgeinit_kernel(const int* __restrict__ edge_index,
                                int* __restrict__ src, int* __restrict__ dst,
                                int* __restrict__ emask) {
  size_t t = (size_t)blockIdx.x * 256 + threadIdx.x;
  if (t >= (size_t)Bb * Ee) return;
  size_t b = t >> 14;             // E = 16384
  size_t e = t & (Ee - 1);
  src[t]   = edge_index[b * (2 * Ee) + e];
  dst[t]   = edge_index[b * (2 * Ee) + Ee + e];
  emask[t] = 1;
}

__global__ void cvt_bf16_kernel(const float* __restrict__ in, unsigned short* __restrict__ out,
                                size_t count) {
  size_t t = (size_t)blockIdx.x * 256 + threadIdx.x;
  if (t >= count) return;
  unsigned int u = __float_as_uint(in[t]);
  unsigned int r = u + 0x7FFFu + ((u >> 16) & 1u);   // round-to-nearest-even
  out[t] = (unsigned short)(r >> 16);
}

// transpose KxO f32 weight [k][o] -> bf16 [o][k] so B fragments are contiguous per lane
__global__ void wtransg_kernel(const float* __restrict__ w, unsigned short* __restrict__ wT,
                               int K, int O) {
  int t = blockIdx.x * 256 + threadIdx.x;
  if (t >= K * O) return;
  int o = t / K, kk = t % K;
  unsigned int u = __float_as_uint(w[kk * O + o]);
  unsigned int r = u + 0x7FFFu + ((u >> 16) & 1u);
  wT[t] = (unsigned short)(r >> 16);
}

// ---------------- edge scatter (message aggregation) ----------------
// one 64-thread block per (b,e); thread = feature channel; atomics resolve in L2
__global__ void scatter_kernel(const float* __restrict__ x, float* __restrict__ agg,
                               const int* __restrict__ src, const int* __restrict__ dst,
                               const int* __restrict__ emask, const int* __restrict__ eaidx,
                               const float* __restrict__ emb, int use_ea, int n) {
  int be = blockIdx.x;                 // b*E + e
  if (!emask[be]) return;
  int b = be >> 14;
  int h = threadIdx.x;                 // 64 threads
  int s = src[be], d = dst[be];
  float v = x[((size_t)b * n + s) * Hh + h];
  if (use_ea) v *= emb[(size_t)eaidx[be] * Hh + h];
  atomicAdd(&agg[((size_t)b * n + d) * Hh + h], v);
}

// ---------------- WMMA conv GEMM: out = relu(agg@wrel + x@wroot + b) ----------------
// block = 128 threads = 4 waves; wave w owns N-tile w (16 of 64 output channels);
// block computes one 16-row M tile for one batch. K=64 => 2 k-steps of 32 (bf16).
__global__ void __launch_bounds__(128)
conv_wmma_kernel(const unsigned short* __restrict__ xb, const unsigned short* __restrict__ aggb,
                 const unsigned short* __restrict__ wrootT, const unsigned short* __restrict__ wrelT,
                 const float* __restrict__ bias, float* __restrict__ out, int n) {
  int mtiles = n >> 4;
  int b     = blockIdx.x / mtiles;
  int mtile = blockIdx.x % mtiles;
  int wave  = threadIdx.x >> 5;        // N tile index 0..3
  int lane  = threadIdx.x & 31;

  int mrow  = mtile * 16 + (lane & 15);
  int h2    = (lane < 16) ? 0 : 4;
  size_t rowbase = ((size_t)b * n + mrow) * Hh;  // bf16 elements

  int ncol = wave * 16 + (lane & 15);
  int bko  = (lane >= 16) ? 16 : 0;

  f32x8 c = {0.f, 0.f, 0.f, 0.f, 0.f, 0.f, 0.f, 0.f};

#pragma unroll
  for (int kb = 0; kb < 2; kb++) {
    bf16x16 aa  = load_a_frag(aggb + rowbase + kb * 32, h2);
    bf16x16 ax  = load_a_frag(xb   + rowbase + kb * 32, h2);
    bf16x16 bre = load_b_frag(wrelT  + ncol * Hh + kb * 32 + bko);
    bf16x16 bro = load_b_frag(wrootT + ncol * Hh + kb * 32 + bko);
    c = __builtin_amdgcn_wmma_f32_16x16x32_bf16(false, aa, false, bre, (short)0, c, false, false);
    c = __builtin_amdgcn_wmma_f32_16x16x32_bf16(false, ax, false, bro, (short)0, c, false, false);
  }

  float bv = bias[ncol];
#pragma unroll
  for (int r = 0; r < 8; r++) {                  // D: VGPR r -> M = r (lanes 0-15) or r+8
    int row = mtile * 16 + r + ((lane >= 16) ? 8 : 0);
    float v = c[r] + bv;
    out[((size_t)b * n + row) * Hh + ncol] = fmaxf(v, 0.f);
  }
}

// ---------------- WMMA dense: out[M][O] = act(in[M][K] @ W[K][O] + b) ----------------
// grid = (M/16) * (O/64); block = 4 waves; wave owns one 16-col N tile of a 64-col group
__global__ void __launch_bounds__(128)
dense_wmma_kernel(const unsigned short* __restrict__ inb,  // [M][K] bf16
                  const unsigned short* __restrict__ wT,   // [O][K] bf16
                  const float* __restrict__ bias, float* __restrict__ out,
                  int M, int K, int O, int relu) {
  int mtiles = M >> 4;
  int mtile = blockIdx.x % mtiles;
  int og    = blockIdx.x / mtiles;
  int wave  = threadIdx.x >> 5;
  int lane  = threadIdx.x & 31;

  int mrow = mtile * 16 + (lane & 15);
  int h2   = (lane < 16) ? 0 : 4;
  int ncol = og * 64 + wave * 16 + (lane & 15);
  int bko  = (lane >= 16) ? 16 : 0;

  f32x8 c = {0.f, 0.f, 0.f, 0.f, 0.f, 0.f, 0.f, 0.f};
  for (int kb = 0; kb < (K >> 5); kb++) {
    bf16x16 a  = load_a_frag(inb + (size_t)mrow * K + kb * 32, h2);
    bf16x16 bf = load_b_frag(wT  + (size_t)ncol * K + kb * 32 + bko);
    c = __builtin_amdgcn_wmma_f32_16x16x32_bf16(false, a, false, bf, (short)0, c, false, false);
  }

  float bv = bias[ncol];
#pragma unroll
  for (int r = 0; r < 8; r++) {
    int row = mtile * 16 + r + ((lane >= 16) ? 8 : 0);
    float v = c[r] + bv;
    if (relu) v = fmaxf(v, 0.f);
    out[(size_t)row * O + ncol] = v;
  }
}

// ---------------- SAG pool: score, top-k, gather, remap, readout ----------------

__global__ void score_kernel(const float* __restrict__ x, const float* __restrict__ agg,
                             const float* __restrict__ swroot, const float* __restrict__ swrel,
                             const float* __restrict__ sb, float* __restrict__ score, int n) {
  int t = blockIdx.x * 256 + threadIdx.x;
  if (t >= Bb * n) return;
  int b = t / n, node = t % n;
  const float* xr = x   + ((size_t)b * n + node) * Hh;
  const float* ar = agg + ((size_t)b * n + node) * Hh;
  float acc = sb[0];
#pragma unroll 4
  for (int k = 0; k < Hh; k++) acc += ar[k] * swrel[k] + xr[k] * swroot[k];
  score[t] = acc;
}

// one block per graph; blockDim == n (power of two); bitonic sort descending, ties by index
__global__ void topk_kernel(const float* __restrict__ score, int* __restrict__ perm,
                            float* __restrict__ vals, int* __restrict__ newid, int n, int k) {
  __shared__ float sv[1024];
  __shared__ int   si[1024];
  int b = blockIdx.x, t = threadIdx.x;
  sv[t] = score[(size_t)b * n + t];
  si[t] = t;
  __syncthreads();
  for (int ks = 2; ks <= n; ks <<= 1) {
    for (int j = ks >> 1; j > 0; j >>= 1) {
      int p = t ^ j;
      if (p > t) {
        bool desc = ((t & ks) == 0);
        float v1 = sv[t], v2 = sv[p];
        int   i1 = si[t], i2 = si[p];
        bool before = (v1 > v2) || (v1 == v2 && i1 < i2);
        if (desc ? !before : before) { sv[t] = v2; sv[p] = v1; si[t] = i2; si[p] = i1; }
      }
      __syncthreads();
    }
  }
  if (t < k) {
    perm[(size_t)b * k + t] = si[t];
    vals[(size_t)b * k + t] = sv[t];
  }
  newid[(size_t)b * n + si[t]] = (t < k) ? t : -1;
}

__global__ void newx_kernel(const float* __restrict__ xin, const int* __restrict__ perm,
                            const float* __restrict__ vals, float* __restrict__ xout, int k) {
  size_t t = (size_t)blockIdx.x * 256 + threadIdx.x;
  if (t >= (size_t)Bb * k * Hh) return;
  int h = (int)(t & 63);
  size_t bj = t >> 6;
  int j = (int)(bj % k);
  int b = (int)(bj / k);
  int p = perm[(size_t)b * k + j];
  float g = tanhf(vals[(size_t)b * k + j]);
  xout[((size_t)b * k + j) * Hh + h] = xin[((size_t)b * (2 * k) + p) * Hh + h] * g;
}

__global__ void remap_kernel(int* __restrict__ src, int* __restrict__ dst,
                             int* __restrict__ emask, const int* __restrict__ newid, int n) {
  int be = blockIdx.x * 256 + threadIdx.x;
  if (be >= Bb * Ee) return;
  int b = be >> 14;
  if (!emask[be]) { src[be] = 0; dst[be] = 0; return; }
  int ns = newid[(size_t)b * n + src[be]];
  int nd = newid[(size_t)b * n + dst[be]];
  int keep = (ns >= 0) && (nd >= 0);
  src[be] = keep ? ns : 0;
  dst[be] = keep ? nd : 0;
  emask[be] = keep;
}

__global__ void readout_kernel(const float* __restrict__ x, float* __restrict__ racc, int k) {
  int b = blockIdx.x, h = threadIdx.x;    // 64 threads
  float m = -INFINITY, s = 0.f;
  for (int j = 0; j < k; j++) {
    float v = x[((size_t)b * k + j) * Hh + h];
    m = fmaxf(m, v);
    s += v;
  }
  racc[b * 128 + h]      += m;
  racc[b * 128 + 64 + h] += s / (float)k;
}

// ---------------- log_softmax ----------------

__global__ void lsm_kernel(const float* __restrict__ in, float* __restrict__ out) {
  __shared__ float red[128];
  int b = blockIdx.x, t = threadIdx.x;   // 128 threads
  float v = in[b * 128 + t];
  red[t] = v;
  __syncthreads();
  for (int s = 64; s > 0; s >>= 1) { if (t < s) red[t] = fmaxf(red[t], red[t + s]); __syncthreads(); }
  float mx = red[0];
  __syncthreads();
  red[t] = expf(v - mx);
  __syncthreads();
  for (int s = 64; s > 0; s >>= 1) { if (t < s) red[t] += red[t + s]; __syncthreads(); }
  out[b * 128 + t] = v - mx - logf(red[0]);
}

// ---------------- host orchestration ----------------

extern "C" void kernel_launch(void* const* d_in, const int* in_sizes, int n_in,
                              void* d_out, int out_size, void* d_ws, size_t ws_size,
                              hipStream_t stream) {
  (void)in_sizes; (void)n_in; (void)out_size; (void)ws_size;

  const int*   x_idx      = (const int*)d_in[0];
  const int*   edge_index = (const int*)d_in[1];
  const int*   eaidx      = (const int*)d_in[2];
  const float* emb        = (const float*)d_in[3];
  auto P = [&](int i) { return (const float*)d_in[i]; };
  // params: 4 + layer*6 : wroot, wrel, b, pool_wroot, pool_wrel, pool_b ; 28..33 lin

  char* ws = (char*)d_ws;
  size_t off = 0;
  auto carve = [&](size_t bytes) -> void* {
    void* p = ws + off;
    off += (bytes + 255) & ~(size_t)255;
    return p;
  };
  const size_t XBYTES = (size_t)Bb * Nn * Hh * 4;
  float*          xA     = (float*)carve(XBYTES);
  float*          xB     = (float*)carve(XBYTES);
  float*          agg    = (float*)carve(XBYTES);
  unsigned short* xb16   = (unsigned short*)carve(XBYTES / 2);
  unsigned short* agg16  = (unsigned short*)carve(XBYTES / 2);
  unsigned short* wrootT = (unsigned short*)carve(Hh * Hh * 2);
  unsigned short* wrelT  = (unsigned short*)carve(Hh * Hh * 2);
  int*   srcW   = (int*)carve((size_t)Bb * Ee * 4);
  int*   dstW   = (int*)carve((size_t)Bb * Ee * 4);
  int*   emaskW = (int*)carve((size_t)Bb * Ee * 4);
  float* score  = (float*)carve((size_t)Bb * Nn * 4);
  int*   perm   = (int*)carve((size_t)Bb * Nn * 4);
  float* valsS  = (float*)carve((size_t)Bb * Nn * 4);
  int*   newid  = (int*)carve((size_t)Bb * Nn * 4);
  float* racc   = (float*)carve(Bb * 128 * 4);
  float* h1     = (float*)carve(Bb * 64 * 4);
  float* h2     = (float*)carve(Bb * 64 * 4);
  float* logits = (float*)carve(Bb * 128 * 4);
  unsigned short* raccb = (unsigned short*)carve(Bb * 128 * 2);
  unsigned short* h1b   = (unsigned short*)carve(Bb * 64 * 2);
  unsigned short* h2b   = (unsigned short*)carve(Bb * 64 * 2);
  unsigned short* w1T   = (unsigned short*)carve(128 * 64 * 2);
  unsigned short* w2T   = (unsigned short*)carve(64 * 64 * 2);
  unsigned short* w3T   = (unsigned short*)carve(64 * 128 * 2);

  // node embeddings + working edge lists + readout accumulator
  embed_kernel<<<(unsigned)(((size_t)Bb * Nn * Hh + 255) / 256), 256, 0, stream>>>(x_idx, emb, xA, (size_t)Bb * Nn);
  edgeinit_kernel<<<(Bb * Ee + 255) / 256, 256, 0, stream>>>(edge_index, srcW, dstW, emaskW);
  fill0_kernel<<<(Bb * 128 + 255) / 256, 256, 0, stream>>>(racc, Bb * 128);

  float* xcur  = xA;   // current node features (n nodes)
  float* xconv = xB;   // conv output
  int n = Nn;

  for (int layer = 0; layer < 4; layer++) {
    const float* wroot  = P(4 + layer * 6 + 0);
    const float* wrel   = P(4 + layer * 6 + 1);
    const float* bconv  = P(4 + layer * 6 + 2);
    const float* pwroot = P(4 + layer * 6 + 3);
    const float* pwrel  = P(4 + layer * 6 + 4);
    const float* pb     = P(4 + layer * 6 + 5);

    size_t xcnt = (size_t)Bb * n * Hh;

    // ---- graph conv ----
    fill0_kernel<<<(unsigned)((xcnt + 255) / 256), 256, 0, stream>>>(agg, xcnt);
    scatter_kernel<<<Bb * Ee, 64, 0, stream>>>(xcur, agg, srcW, dstW, emaskW, eaidx, emb, 1, n);
    cvt_bf16_kernel<<<(unsigned)((xcnt + 255) / 256), 256, 0, stream>>>(xcur, xb16, xcnt);
    cvt_bf16_kernel<<<(unsigned)((xcnt + 255) / 256), 256, 0, stream>>>(agg, agg16, xcnt);
    wtransg_kernel<<<(Hh * Hh + 255) / 256, 256, 0, stream>>>(wroot, wrootT, Hh, Hh);
    wtransg_kernel<<<(Hh * Hh + 255) / 256, 256, 0, stream>>>(wrel, wrelT, Hh, Hh);
    conv_wmma_kernel<<<Bb * (n / 16), 128, 0, stream>>>(xb16, agg16, wrootT, wrelT, bconv, xconv, n);

    // ---- SAG pool ----
    fill0_kernel<<<(unsigned)((xcnt + 255) / 256), 256, 0, stream>>>(agg, xcnt);
    scatter_kernel<<<Bb * Ee, 64, 0, stream>>>(xconv, agg, srcW, dstW, emaskW, eaidx, emb, 0, n);
    score_kernel<<<(Bb * n + 255) / 256, 256, 0, stream>>>(xconv, agg, pwroot, pwrel, pb, score, n);
    int k = n / 2;
    topk_kernel<<<Bb, n, 0, stream>>>(score, perm, valsS, newid, n, k);
    newx_kernel<<<(unsigned)(((size_t)Bb * k * Hh + 255) / 256), 256, 0, stream>>>(xconv, perm, valsS, xcur, k);
    remap_kernel<<<(Bb * Ee + 255) / 256, 256, 0, stream>>>(srcW, dstW, emaskW, newid, n);
    readout_kernel<<<Bb, 64, 0, stream>>>(xcur, racc, k);
    n = k;
  }

  // ---- final MLP on the WMMA pipe (bf16 operands, f32 accumulation) ----
  cvt_bf16_kernel<<<(Bb * 128 + 255) / 256, 256, 0, stream>>>(racc, raccb, Bb * 128);
  wtransg_kernel<<<(128 * 64 + 255) / 256, 256, 0, stream>>>(P(28), w1T, 128, 64);
  dense_wmma_kernel<<<4, 128, 0, stream>>>(raccb, w1T, P(29), h1, 64, 128, 64, 1);

  cvt_bf16_kernel<<<(Bb * 64 + 255) / 256, 256, 0, stream>>>(h1, h1b, Bb * 64);
  wtransg_kernel<<<(64 * 64 + 255) / 256, 256, 0, stream>>>(P(30), w2T, 64, 64);
  dense_wmma_kernel<<<4, 128, 0, stream>>>(h1b, w2T, P(31), h2, 64, 64, 64, 1);

  cvt_bf16_kernel<<<(Bb * 64 + 255) / 256, 256, 0, stream>>>(h2, h2b, Bb * 64);
  wtransg_kernel<<<(64 * 128 + 255) / 256, 256, 0, stream>>>(P(32), w3T, 64, 128);
  dense_wmma_kernel<<<8, 128, 0, stream>>>(h2b, w3T, P(33), logits, 64, 64, 128, 0);

  lsm_kernel<<<Bb, 128, 0, stream>>>(logits, (float*)d_out);
}
